// GTSCNN_38156489458302
// MI455X (gfx1250) — compile-verified, matrix operand order checked by reading, not run on hardware
//
#include <hip/hip_runtime.h>

typedef __attribute__((ext_vector_type(16))) _Float16 v16h;
typedef __attribute__((ext_vector_type(16))) __bf16   v16bf;
typedef __attribute__((ext_vector_type(8)))  float    v8f;
typedef __attribute__((ext_vector_type(4)))  unsigned uv4;

#define B_    4
#define L_    256
#define E_    256
#define F_    256
#define K3F   768
#define LPAD  259   // 1 front halo + 256 + 2 back halo

union Frag {
  uv4   q[2];
  v16h  h;
  v16bf b;
};

__device__ __forceinline__ unsigned f2bf(float f) {
  unsigned u = __builtin_bit_cast(unsigned, f);
  return (u + 0x7FFFu + ((u >> 16) & 1u)) >> 16;   // RNE to bf16 bits
}
__device__ __forceinline__ float bf2f(unsigned s) {
  return __builtin_bit_cast(float, s << 16);
}
__device__ __forceinline__ unsigned short f2h(float f) {
  _Float16 h = (_Float16)f;
  return __builtin_bit_cast(unsigned short, h);
}
// packed bf16 pair multiply: a*s elementwise (RNE repack)
__device__ __forceinline__ unsigned bfpair_mul(unsigned a, unsigned s) {
  float lo = bf2f(a & 0xFFFFu) * bf2f(s & 0xFFFFu);
  float hi = bf2f(a >> 16)     * bf2f(s >> 16);
  return f2bf(lo) | (f2bf(hi) << 16);
}

// Load a 16x32 (f16/bf16) fragment: row-major matrix, one row per lane (m = lane&15).
// Lanes 0-15 take K = kb+0..7 and kb+16..23; lanes 16-31 take kb+8..15 and kb+24..31.
__device__ __forceinline__ Frag load_frag(const unsigned short* __restrict__ row,
                                          int kb, int hi8) {
  Frag f;
  f.q[0] = *(const uv4*)(row + kb + hi8);
  f.q[1] = *(const uv4*)(row + kb + hi8 + 16);
  return f;
}

// ---------------------------------------------------------------------------
// K1: embedding gather -> f16 Xpad ; conv weights -> tap-major f16 Wt ;
//     triple_w/score_w -> bf16 cw[7][768]
// ---------------------------------------------------------------------------
__global__ __launch_bounds__(256)
void prepare_kernel(const int* __restrict__ ids,
                    const float* __restrict__ emb,
                    const float* __restrict__ w2,
                    const float* __restrict__ w3,
                    const float* __restrict__ w4,
                    const float* __restrict__ triple_w,
                    const float* __restrict__ score_w,
                    unsigned short* __restrict__ Xpad,   // f16  [B][259][256]
                    unsigned short* __restrict__ Wt,     // f16  [9][256][256]
                    unsigned short* __restrict__ cw)     // bf16 [7][768]
{
  int blk = blockIdx.x, tid = threadIdx.x;
  if (blk < B_ * LPAD) {                       // 1036 blocks: Xpad rows
    int b = blk / LPAD, lr = blk % LPAD;
    int l = lr - 1;
    float v = 0.f;
    if (l >= 0 && l < L_) {
      int id = ids[b * L_ + l];
      v = emb[(size_t)id * E_ + tid];
    }
    Xpad[(size_t)blk * E_ + tid] = f2h(v);
  } else if (blk < B_ * LPAD + 9 * F_) {       // 2304 blocks: weight rows
    int t  = blk - B_ * LPAD;
    int tg = t / F_;                           // global tap 0..8
    int f  = t % F_;
    float v;
    if (tg < 2)      v = w2[(size_t)f * (E_ * 2) + tid * 2 + tg];
    else if (tg < 5) v = w3[(size_t)f * (E_ * 3) + tid * 3 + (tg - 2)];
    else             v = w4[(size_t)f * (E_ * 4) + tid * 4 + (tg - 5)];
    Wt[((size_t)tg * F_ + f) * E_ + tid] = f2h(v);
  } else {                                     // 7 blocks: cw rows
    int c = blk - (B_ * LPAD + 9 * F_);
    for (int k = tid; k < K3F; k += 256) {
      float v = (c < 6) ? triple_w[c * K3F + k] : score_w[k];
      cw[c * K3F + k] = (unsigned short)f2bf(v);
    }
  }
}

// ---------------------------------------------------------------------------
// K2: conv-as-GEMM, j-blocked. One wave computes a 16x64 tile of out[b]:
// A fragment (shifted activations) reused across 4 feature tiles per WMMA step.
// Y = sum_t Shift_t(Xpad) @ Wt[t]^T via v_wmma_f32_16x16x32_f16.
// 768 wave-tiles: b(4) x cid(3) x fgroup(4) x ltile(16).
// ---------------------------------------------------------------------------
__global__ __launch_bounds__(256)
void conv_wmma_kernel(const unsigned short* __restrict__ Xpad,
                      const unsigned short* __restrict__ Wt,
                      const float* __restrict__ b2,
                      const float* __restrict__ b3,
                      const float* __restrict__ b4,
                      unsigned short* __restrict__ out_bf) // bf16 [B][256][768]
{
  int wg   = blockIdx.x * 8 + (threadIdx.x >> 5);   // 768 tiles
  int lane = threadIdx.x & 31;
  int b    = wg / 192;
  int rem  = wg % 192;
  int cid  = rem / 64;                 // conv id 0..2 (k=2,3,4)
  int rem2 = rem % 64;
  int fg   = rem2 >> 4;                // feature group: 64 features
  int l0   = (rem2 & 15) << 4;         // row tile base
  int f0   = fg << 6;                  // feature base within conv

  int ntaps = (cid == 0) ? 2 : ((cid == 1) ? 3 : 4);
  int pad   = (cid == 0) ? 0 : 1;
  int tapb  = (cid == 0) ? 0 : ((cid == 1) ? 2 : 5);

  int m   = lane & 15;
  int hi8 = (lane >> 4) << 3;

  v8f acc[4];
#pragma unroll
  for (int j = 0; j < 4; ++j) acc[j] = (v8f){0.f,0.f,0.f,0.f,0.f,0.f,0.f,0.f};

  for (int ti = 0; ti < ntaps; ++ti) {
    const unsigned short* Arow =
        Xpad + (size_t)b * LPAD * E_ + (size_t)(1 + l0 + (ti - pad) + m) * E_;
    const unsigned short* Wtap = Wt + (size_t)(tapb + ti) * F_ * E_;
#pragma unroll
    for (int ks = 0; ks < 8; ++ks) {
      int kb = ks * 32;
      Frag fa = load_frag(Arow, kb, hi8);
#pragma unroll
      for (int j = 0; j < 4; ++j) {
        Frag fb = load_frag(Wtap + (size_t)(f0 + j * 16 + m) * E_, kb, hi8);
        acc[j] = __builtin_amdgcn_wmma_f32_16x16x32_f16(
            false, fa.h, false, fb.h, (short)0, acc[j], false, false);
      }
    }
  }

  const float* bias = (cid == 0) ? b2 : ((cid == 1) ? b3 : b4);
  int rb = (lane >> 4) * 8;            // hi half-lanes hold M+8
#pragma unroll
  for (int j = 0; j < 4; ++j) {
    int fl  = f0 + j * 16 + m;         // feature within conv
    float bv = bias[fl];
    int col  = cid * F_ + fl;          // global feature 0..767
#pragma unroll
    for (int r = 0; r < 8; ++r) {
      int l = l0 + r + rb;
      float v = acc[j][r] + bv;
      out_bf[((size_t)(b * L_ + l)) * K3F + col] = (unsigned short)f2bf(v);
    }
  }
}

// ---------------------------------------------------------------------------
// K3: P_c[b] = (out ⊙ w_c) @ out^T per (b,c), fused diag-scale + j-blocked.
// One wave computes a 16x64 (i,j) tile: the scaled A fragment is built once
// per k-step and reused across 4 WMMAs. 1792 wave-tiles:
// b(4) x c(7) x itile(16) x jgroup(4); K=768 in 24 steps.
// ---------------------------------------------------------------------------
__global__ __launch_bounds__(256)
void pair_wmma_kernel(const unsigned short* __restrict__ out_bf, // bf16 [B][256][768]
                      const unsigned short* __restrict__ cw,     // bf16 [7][768]
                      const float* __restrict__ triple_b,
                      const float* __restrict__ score_b,
                      float* __restrict__ dout)
{
  int wg   = blockIdx.x * 8 + (threadIdx.x >> 5);  // 1792 tiles
  int lane = threadIdx.x & 31;
  int b    = wg / 448;
  int rem  = wg % 448;
  int c    = rem / 64;             // 0..6 (6 == score head)
  int rem2 = rem % 64;
  int i0   = (rem2 >> 2) << 4;     // i tile base
  int jg   = rem2 & 3;
  int j0   = jg << 6;              // j group base (64 columns)

  int m   = lane & 15;
  int hi8 = (lane >> 4) << 3;

  const unsigned short* Arow = out_bf + ((size_t)(b * L_ + (i0 + m))) * K3F;
  const unsigned short* Crow = cw + (size_t)c * K3F;
  const unsigned short* Bbase = out_bf + ((size_t)b * L_) * K3F;

  v8f acc[4];
#pragma unroll
  for (int j = 0; j < 4; ++j) acc[j] = (v8f){0.f,0.f,0.f,0.f,0.f,0.f,0.f,0.f};

#pragma unroll 2
  for (int kb = 0; kb < K3F; kb += 32) {
    Frag fa = load_frag(Arow, kb, hi8);
    Frag fc = load_frag(Crow, kb, hi8);
#pragma unroll
    for (int d = 0; d < 2; ++d) {      // fuse diag(w_c) into A fragment
      fa.q[d].x = bfpair_mul(fa.q[d].x, fc.q[d].x);
      fa.q[d].y = bfpair_mul(fa.q[d].y, fc.q[d].y);
      fa.q[d].z = bfpair_mul(fa.q[d].z, fc.q[d].z);
      fa.q[d].w = bfpair_mul(fa.q[d].w, fc.q[d].w);
    }
#pragma unroll
    for (int j = 0; j < 4; ++j) {
      Frag fb = load_frag(Bbase + (size_t)(j0 + j * 16 + m) * K3F, kb, hi8);
      acc[j] = __builtin_amdgcn_wmma_f32_16x16x32_bf16(
          false, fa.b, false, fb.b, (short)0, acc[j], false, false);
    }
  }

  float bias = (c < 6) ? triple_b[c] : score_b[0];
  int rb = (lane >> 4) * 8;
  const size_t TL = (size_t)B_ * L_ * L_ * 6;
#pragma unroll
  for (int j = 0; j < 4; ++j) {
    int jj = j0 + j * 16 + m;
#pragma unroll
    for (int r = 0; r < 8; ++r) {
      int i = i0 + r + rb;
      float v = acc[j][r] + bias;
      if (c < 6)
        dout[(((size_t)b * L_ + i) * L_ + jj) * 6 + c] = v;
      else
        dout[TL + ((size_t)b * L_ + i) * L_ + jj] = v;
    }
  }
}

// ---------------------------------------------------------------------------
extern "C" void kernel_launch(void* const* d_in, const int* in_sizes, int n_in,
                              void* d_out, int out_size, void* d_ws, size_t ws_size,
                              hipStream_t stream) {
  (void)in_sizes; (void)n_in; (void)out_size; (void)ws_size;

  const int*   ids      = (const int*)  d_in[0];
  const float* emb      = (const float*)d_in[1];
  const float* w2       = (const float*)d_in[2];
  const float* b2       = (const float*)d_in[3];
  const float* w3       = (const float*)d_in[4];
  const float* b3       = (const float*)d_in[5];
  const float* w4       = (const float*)d_in[6];
  const float* b4       = (const float*)d_in[7];
  const float* triple_w = (const float*)d_in[8];
  const float* triple_b = (const float*)d_in[9];
  const float* score_w  = (const float*)d_in[10];
  const float* score_b  = (const float*)d_in[11];
  float* dout = (float*)d_out;

  char* ws = (char*)d_ws;
  unsigned short* Xpad   = (unsigned short*)(ws + 0);          //   530,432 B
  unsigned short* Wt     = (unsigned short*)(ws + 530432);     // 1,179,648 B
  unsigned short* out_bf = (unsigned short*)(ws + 1710080);    // 1,572,864 B
  unsigned short* cw     = (unsigned short*)(ws + 3282944);    //    10,752 B

  // K1: gather + pack (1036 Xpad rows + 2304 weight rows + 7 cw rows)
  prepare_kernel<<<B_ * LPAD + 9 * F_ + 7, 256, 0, stream>>>(
      ids, emb, w2, w3, w4, triple_w, score_w, Xpad, Wt, cw);

  // K2: conv GEMM, 16x64 tiles (768 wave-tiles, 8 waves/block)
  conv_wmma_kernel<<<96, 256, 0, stream>>>(Xpad, Wt, b2, b3, b4, out_bf);

  // K3: 7 fused diag-scaled Gram GEMMs per batch, 16x64 tiles (1792 wave-tiles)
  pair_wmma_kernel<<<224, 256, 0, stream>>>(out_bf, cw, triple_b, score_b, dout);
}
